// Encoder_19988777796282
// MI455X (gfx1250) — matrix operand order repaired and checked by reading, not compiled, not run on hardware
//
#include <hip/hip_runtime.h>
#include <stdint.h>

// Problem constants (from reference)
#define B_N   65536
#define T_N   30
#define V_N   21
#define H_N   64
#define D_N   784
#define EOS_N 20
#define VP    32            // vocab padded to 32 (WMMA K / N tile granularity)
#define NT    12            // N tiles for 3H = 192
#define KC_D  25            // encoder K chunks: 800/32 (784 padded)

typedef __attribute__((ext_vector_type(16))) _Float16 v16h;
typedef __attribute__((ext_vector_type(8)))  float    v8f;

union AFrag { v16h h; uint32_t u[8]; };

// Wave-local LDS fence: DS ops are in-order per wave; this guarantees prior
// ds_stores are visible to subsequent ds_loads (CDNA5 split counter wait).
__device__ __forceinline__ void lds_fence() {
  asm volatile("s_wait_dscnt 0" ::: "memory");
}

__device__ __forceinline__ uint32_t pack2h(float a, float b) {
  union { _Float16 f; uint16_t q; } ua, ub;
  ua.f = (_Float16)a; ub.f = (_Float16)b;
  return (uint32_t)ua.q | ((uint32_t)ub.q << 16);
}

__device__ __forceinline__ v8f wmma_f16(v16h a, v16h b, v8f c) {
  // 8 args: (neg_a, A, neg_b, B, c_mod, C, reuse_a, reuse_b)
  return __builtin_amdgcn_wmma_f32_16x16x32_f16(false, a, false, b, (short)0, c,
                                                false, false);
}

// B fragment (K=32 x N=16, f16): LDS pre-packed as [kpair(16)][n(16)] dwords,
// dword = {f16(k=2kp,n), f16(k=2kp+1,n)}. Lane L -> N=L%16, g=L/16 covers
// K in [16g,16g+16), dword v -> kpair = 8g+v.
__device__ __forceinline__ v16h loadB(const uint32_t* bst, int lane) {
  const int n = lane & 15, g = lane >> 4;
  AFrag B;
#pragma unroll
  for (int v = 0; v < 8; ++v) B.u[v] = bst[(g * 8 + v) * 16 + n];
  return B.h;
}

// A fragment (M=16 x K=32, f16) from f16 LDS stored element-wise [16][stride].
// Lane L -> M=L%16; g=L/16; dword v -> K=16g+2v.
// k is always even so each dword is one aligned b32 load of an f16 pair.
__device__ __forceinline__ v16h buildA16(const _Float16* act, int stride,
                                         int kbase, int lane) {
  const int m = lane & 15, g = lane >> 4;
  AFrag A;
#pragma unroll
  for (int v = 0; v < 8; ++v) {
    const int k = kbase + g * 16 + 2 * v;
    A.u[v] = *(const uint32_t*)(act + m * stride + k);
  }
  return A.h;
}

__device__ __forceinline__ v8f splat8(float x) {
  v8f r;
#pragma unroll
  for (int i = 0; i < 8; ++i) r[i] = x;
  return r;
}

__device__ __forceinline__ float sigmoidf_(float x) {
  return 1.0f / (1.0f + __expf(-x));
}
__device__ __forceinline__ float tanhf_(float x) {
  x = fminf(fmaxf(x, -15.0f), 15.0f);
  const float e = __expf(-2.0f * x);
  return (1.0f - e) / (1.0f + e);
}

// ---------------------------------------------------------------------------
// Kernel 0: zero the scalar entropy accumulator slot in d_out.
// ---------------------------------------------------------------------------
__global__ void zero_ent_kernel(float* __restrict__ out) {
  out[(size_t)B_N * T_N * V_N + B_N] = 0.0f;
}

// ---------------------------------------------------------------------------
// Kernel 1: encoder  h0 = ELU(x @ W_enc^T + b_enc)   [B,64] -> d_ws
// 256 threads = 8 waves, each wave = 16 batch rows, WG = 128 rows.
// W_enc pre-packed to f16 B-fragments in LDS; x staged per-wave as f16.
// ---------------------------------------------------------------------------
__global__ void enc_kernel(const float* __restrict__ x,
                           const float* __restrict__ Wenc,
                           const float* __restrict__ benc,
                           float* __restrict__ h0) {
  extern __shared__ char smem[];
  uint32_t*  BW = (uint32_t*)smem;                                // [25][4][16][16]
  _Float16*  xT = (_Float16*)(smem + (size_t)KC_D * 4 * 256 * 4); // [8][16][32] f16

  const int tid  = threadIdx.x;
  const int lane = tid & 31;
  const int wave = tid >> 5;

  // Pack W_enc^T into B-fragment layout (zero-pad K 784..799).
  for (int idx = tid; idx < KC_D * 4 * 256; idx += 256) {
    const int c    = idx >> 10;
    const int tile = (idx >> 8) & 3;
    const int kp   = (idx >> 4) & 15;
    const int n    = idx & 15;
    const int k    = c * 32 + 2 * kp;
    const int ng   = tile * 16 + n;
    const float a = (k     < D_N) ? Wenc[(size_t)ng * D_N + k]     : 0.0f;
    const float b = (k + 1 < D_N) ? Wenc[(size_t)ng * D_N + k + 1] : 0.0f;
    BW[idx] = pack2h(a, b);
  }
  __syncthreads();

  const int rowbase = blockIdx.x * 128 + wave * 16;
  _Float16* xw = xT + wave * (16 * 32);
  const int n16 = lane & 15, g = lane >> 4;
  const int row = lane >> 1, cb = (lane & 1) * 16;  // staging assignment
  const float* xRow = x + (size_t)(rowbase + row) * D_N;

  v8f acc[4];
#pragma unroll
  for (int t = 0; t < 4; ++t) acc[t] = splat8(benc[t * 16 + n16]);

  // Main chunks 0..23: branch-free staging.
  for (int c = 0; c < KC_D - 1; ++c) {
    lds_fence();  // previous chunk's A-frag loads complete before overwrite
#pragma unroll
    for (int i = 0; i < 4; ++i) {
      const float4 v4 = *(const float4*)(xRow + c * 32 + cb + i * 4);
      uint2 pk;
      pk.x = pack2h(v4.x, v4.y);
      pk.y = pack2h(v4.z, v4.w);
      *(uint2*)(xw + row * 32 + cb + i * 4) = pk;  // ds_store_b64
    }
    lds_fence();
    const v16h aX = buildA16(xw, 32, 0, lane);
    v16h b0 = loadB(BW + ((c * 4 + 0) << 8), lane);
    v16h b1 = loadB(BW + ((c * 4 + 1) << 8), lane);
    v16h b2 = loadB(BW + ((c * 4 + 2) << 8), lane);
    v16h b3 = loadB(BW + ((c * 4 + 3) << 8), lane);
    acc[0] = wmma_f16(aX, b0, acc[0]);
    acc[1] = wmma_f16(aX, b1, acc[1]);
    acc[2] = wmma_f16(aX, b2, acc[2]);
    acc[3] = wmma_f16(aX, b3, acc[3]);
  }
  // Peeled tail chunk 24: cols 768..783 valid, 784..799 zero.
  {
    const int c = KC_D - 1;
    lds_fence();
#pragma unroll
    for (int i = 0; i < 4; ++i) {
      uint2 pk;
      if (cb == 0) {
        const float4 v4 = *(const float4*)(xRow + c * 32 + i * 4);
        pk.x = pack2h(v4.x, v4.y);
        pk.y = pack2h(v4.z, v4.w);
      } else {
        pk.x = 0u; pk.y = 0u;
      }
      *(uint2*)(xw + row * 32 + cb + i * 4) = pk;
    }
    lds_fence();
    const v16h aX = buildA16(xw, 32, 0, lane);
    v16h b0 = loadB(BW + ((c * 4 + 0) << 8), lane);
    v16h b1 = loadB(BW + ((c * 4 + 1) << 8), lane);
    v16h b2 = loadB(BW + ((c * 4 + 2) << 8), lane);
    v16h b3 = loadB(BW + ((c * 4 + 3) << 8), lane);
    acc[0] = wmma_f16(aX, b0, acc[0]);
    acc[1] = wmma_f16(aX, b1, acc[1]);
    acc[2] = wmma_f16(aX, b2, acc[2]);
    acc[3] = wmma_f16(aX, b3, acc[3]);
  }

#pragma unroll
  for (int t = 0; t < 4; ++t) {
#pragma unroll
    for (int v = 0; v < 8; ++v) {
      const float val = acc[t][v];
      const float e   = (val > 0.0f) ? val : (__expf(val) - 1.0f);
      const int   m   = v + 8 * g;
      h0[(size_t)(rowbase + m) * H_N + t * 16 + n16] = e;
    }
  }
}

// ---------------------------------------------------------------------------
// Kernel 2: fused 30-step GRU + projection + gumbel-softmax head.
// 256 threads = 8 waves; each wave privately owns 16 batch rows. h state in
// f32 registers, f16 shadow in LDS for WMMA A-fragments. No cross-wave
// barriers inside the T-loop; only wave-local s_wait_dscnt fences.
// WMMAs grouped by A operand so B-fragment LDS latency overlaps matrix ops.
// ---------------------------------------------------------------------------
__global__ void gru_kernel(const float* __restrict__ h0,
                           const float* __restrict__ u,
                           const float* __restrict__ tau_p,
                           const float* __restrict__ Wih,
                           const float* __restrict__ Whh,
                           const float* __restrict__ bih,
                           const float* __restrict__ bhh,
                           const float* __restrict__ initEmb,
                           const float* __restrict__ Wproj,
                           const float* __restrict__ bproj,
                           float* __restrict__ out) {
  extern __shared__ char smem[];
  // Byte offsets (all 8B aligned)
  uint32_t* Bih  = (uint32_t*)(smem);                  // 12288 B
  uint32_t* Bhh  = (uint32_t*)(smem + 12288);          // 24576 B
  uint32_t* Bpj  = (uint32_t*)(smem + 36864);          //  4096 B
  float*    brz  = (float*)(smem + 40960);             //   512 B
  float*    bin  = (float*)(smem + 41472);             //   256 B
  float*    bhn  = (float*)(smem + 41728);             //   256 B
  float*    bpj  = (float*)(smem + 41984);             //   128 B
  _Float16* hH   = (_Float16*)(smem + 42112);          // 16384 B  [8][16][64] f16
  _Float16* inpH = (_Float16*)(smem + 58496);          //  8192 B  [8][16][32] f16
  float*    inpF = (float*)(smem + 66688);             // 16384 B  [8][16][32] f32
  float*    gL   = (float*)(smem + 83072);             // 10752 B  [8][336] gumbel
  // total 93824 B

  const int tid  = threadIdx.x;
  const int lane = tid & 31;
  const int wave = tid >> 5;
  const int rowbase = blockIdx.x * 128;

  // ---- pack weights into B-fragment LDS layouts ----
  for (int idx = tid; idx < NT * 256; idx += 256) {        // W_ih^T, K pad 21->32
    const int tile = idx >> 8, kp = (idx >> 4) & 15, n = idx & 15;
    const int k = 2 * kp, ng = tile * 16 + n;
    const float a = (k     < V_N) ? Wih[ng * V_N + k]     : 0.0f;
    const float b = (k + 1 < V_N) ? Wih[ng * V_N + k + 1] : 0.0f;
    Bih[idx] = pack2h(a, b);
  }
  for (int idx = tid; idx < NT * 2 * 256; idx += 256) {    // W_hh^T (K=64)
    const int tile = idx >> 9, ch = (idx >> 8) & 1, kp = (idx >> 4) & 15, n = idx & 15;
    const int k = ch * 32 + 2 * kp, ng = tile * 16 + n;
    Bhh[idx] = pack2h(Whh[ng * H_N + k], Whh[ng * H_N + k + 1]);
  }
  for (int idx = tid; idx < 4 * 256; idx += 256) {         // W_proj^T, N pad 21->32
    const int tile = idx >> 9, ch = (idx >> 8) & 1, kp = (idx >> 4) & 15, n = idx & 15;
    const int k = ch * 32 + 2 * kp, ng = tile * 16 + n;
    const float a = (ng < V_N) ? Wproj[ng * H_N + k]     : 0.0f;
    const float b = (ng < V_N) ? Wproj[ng * H_N + k + 1] : 0.0f;
    Bpj[idx] = pack2h(a, b);
  }
  for (int j = tid; j < 128; j += 256) brz[j] = bih[j] + bhh[j];
  for (int j = tid; j < 64; j += 256) { bin[j] = bih[128 + j]; bhn[j] = bhh[128 + j]; }
  for (int j = tid; j < 32; j += 256) bpj[j] = (j < V_N) ? bproj[j] : 0.0f;

  // ---- init state: f16 shadow of h0, f16 init_emb input ----
  for (int idx = tid; idx < 128 * H_N; idx += 256)
    hH[idx] = (_Float16)h0[(size_t)rowbase * H_N + idx];
  for (int idx = tid; idx < 128 * VP; idx += 256) {
    const int j = idx & (VP - 1);
    inpH[idx] = (_Float16)((j < V_N) ? initEmb[j] : 0.0f);
  }
  __syncthreads();

  _Float16* hHW   = hH   + wave * 16 * H_N;
  _Float16* inpHW = inpH + wave * 16 * VP;
  float*    inpFW = inpF + wave * 16 * VP;
  float*    gW    = gL   + wave * (16 * V_N);
  const int n16 = lane & 15, g = lane >> 4;
  const int rowWaveBase = rowbase + wave * 16;

  // persistent f32 h in C-matrix layout: hreg[t][v] = h[v+8g][16t+n16]
  v8f hreg[4];
#pragma unroll
  for (int t = 0; t < 4; ++t)
#pragma unroll
    for (int v = 0; v < 8; ++v)
      hreg[t][v] = h0[(size_t)(rowWaveBase + v + 8 * g) * H_N + t * 16 + n16];

  const float tau  = tau_p[0];
  const float itau = 1.0f / tau;
  float entAcc = 0.0f;
  int   len    = T_N;
  bool  fin    = false;

  // W_proj fragments are tiny and loop-invariant: keep them in registers.
  v16h bP[4];
#pragma unroll
  for (int i = 0; i < 4; ++i) bP[i] = loadB(Bpj + i * 256, lane);

  // A-fragments of (old) h; reused from the logits matmul of the prior step.
  v16h aH0 = buildA16(hHW, H_N, 0, lane);
  v16h aH1 = buildA16(hHW, H_N, 32, lane);

  for (int t = 0; t < T_N; ++t) {
    const v16h aI = buildA16(inpHW, VP, 0, lane);  // fed-back logits (pad = 0)

#pragma unroll
    for (int ct = 0; ct < 4; ++ct) {  // column groups of H (16 cols each)
      v8f accR  = splat8(brz[ct * 16 + n16]);
      v8f accZ  = splat8(brz[64 + ct * 16 + n16]);
      v8f accIN = splat8(bin[ct * 16 + n16]);
      v8f accHN = splat8(bhn[ct * 16 + n16]);

      // trio 1: A = aI (gi contributions)
      v16h bR = loadB(Bih + ct * 256, lane);
      v16h bZ = loadB(Bih + (4 + ct) * 256, lane);
      v16h bN = loadB(Bih + (8 + ct) * 256, lane);
      accR  = wmma_f16(aI, bR, accR);
      accZ  = wmma_f16(aI, bZ, accZ);
      accIN = wmma_f16(aI, bN, accIN);

      // trio 2: A = aH0 (gh, K chunk 0)
      bR = loadB(Bhh + (ct * 2 + 0) * 256, lane);
      bZ = loadB(Bhh + ((4 + ct) * 2 + 0) * 256, lane);
      bN = loadB(Bhh + ((8 + ct) * 2 + 0) * 256, lane);
      accR  = wmma_f16(aH0, bR, accR);
      accZ  = wmma_f16(aH0, bZ, accZ);
      accHN = wmma_f16(aH0, bN, accHN);

      // trio 3: A = aH1 (gh, K chunk 1)
      bR = loadB(Bhh + (ct * 2 + 1) * 256, lane);
      bZ = loadB(Bhh + ((4 + ct) * 2 + 1) * 256, lane);
      bN = loadB(Bhh + ((8 + ct) * 2 + 1) * 256, lane);
      accR  = wmma_f16(aH1, bR, accR);
      accZ  = wmma_f16(aH1, bZ, accZ);
      accHN = wmma_f16(aH1, bN, accHN);

#pragma unroll
      for (int v = 0; v < 8; ++v) {  // GRU elementwise in f32
        const float r  = sigmoidf_(accR[v]);
        const float z  = sigmoidf_(accZ[v]);
        const float nn = tanhf_(accIN[v] + r * accHN[v]);
        const float hn = (1.0f - z) * nn + z * hreg[ct][v];
        hreg[ct][v] = hn;
        hHW[(v + 8 * g) * H_N + ct * 16 + n16] = (_Float16)hn;  // f16 shadow
      }
    }
    lds_fence();

    // A-frags of new h; used for logits now and as gate inputs next step.
    aH0 = buildA16(hHW, H_N, 0, lane);
    aH1 = buildA16(hHW, H_N, 32, lane);

    // logits = h_new @ W_proj^T + b_proj: all-register B, no LDS wait.
    {
      v8f accL0 = splat8(bpj[n16]);
      v8f accL1 = splat8(bpj[16 + n16]);
      accL0 = wmma_f16(aH0, bP[0], accL0);
      accL1 = wmma_f16(aH0, bP[2], accL1);
      accL0 = wmma_f16(aH1, bP[1], accL0);
      accL1 = wmma_f16(aH1, bP[3], accL1);
#pragma unroll
      for (int v = 0; v < 8; ++v) {
        const int m = v + 8 * g;
        inpFW[m * VP + n16]      = accL0[v];
        inpHW[m * VP + n16]      = (_Float16)accL0[v];
        inpFW[m * VP + 16 + n16] = accL1[v];
        inpHW[m * VP + 16 + n16] = (_Float16)accL1[v];
      }
    }

    // Stage gumbel noise coalesced with the full wave; prefetch next slice.
    {
      const float* uBase = u + ((size_t)t * B_N + rowWaveBase) * V_N;
      for (int i = lane; i < 16 * V_N; i += 32) {
        const float uu = uBase[i];
        gW[i] = -__logf(-__logf(uu + 1e-10f) + 1e-10f);
      }
      if (t + 1 < T_N)
        __builtin_prefetch(u + ((size_t)(t + 1) * B_N + rowWaveBase) * V_N, 0, 1);
    }
    lds_fence();

    // Gumbel-softmax head: 2 lanes per row, pair reductions via shfl_xor.
    {
      const int m   = lane >> 1;
      const int sub = lane & 1;
      const int jb  = sub ? 11 : 0;
      const int jn  = sub ? 10 : 11;  // sub0: j 0..10, sub1: j 11..20

      float lg[11];
      float mx = -1e30f;
#pragma unroll
      for (int q = 0; q < 11; ++q) {
        const float v = (q < jn) ? inpFW[m * VP + jb + q] : -1e30f;
        lg[q] = v;
        mx = fmaxf(mx, v);
      }
      mx = fmaxf(mx, __shfl_xor(mx, 1, 32));

      float se = 0.0f;
#pragma unroll
      for (int q = 0; q < 11; ++q)
        if (q < jn) se += __expf(lg[q] - mx);
      se += __shfl_xor(se, 1, 32);
      const float lse = __logf(se);

      float entl = 0.0f;
#pragma unroll
      for (int q = 0; q < 11; ++q)
        if (q < jn) { const float lp = lg[q] - mx - lse; entl -= __expf(lp) * lp; }
      entl += __shfl_xor(entl, 1, 32);
      entAcc += entl;  // both lanes of the pair hold the full row entropy

      float av[11];
      float amax = -1e30f;
      int   aidx = 0x7fffffff;
#pragma unroll
      for (int q = 0; q < 11; ++q) {
        if (q < jn) {
          const float val = (lg[q] + gW[m * V_N + jb + q]) * itau;
          av[q] = val;
          if (val > amax) { amax = val; aidx = jb + q; }  // first-max per lane
        }
      }
      {  // combine pair: on exact tie the smaller index wins (jnp.argmax)
        const float oa = __shfl_xor(amax, 1, 32);
        const int   oi = __shfl_xor(aidx, 1, 32);
        if (oa > amax || (oa == amax && oi < aidx)) { amax = oa; aidx = oi; }
      }
      float ss = 0.0f;
#pragma unroll
      for (int q = 0; q < 11; ++q)
        if (q < jn) { const float e = __expf(av[q] - amax); av[q] = e; ss += e; }
      ss += __shfl_xor(ss, 1, 32);
      const float inv = 1.0f / ss;

      float* oRow = out + ((size_t)(rowWaveBase + m) * T_N + t) * V_N;
#pragma unroll
      for (int q = 0; q < 11; ++q) {
        if (q < jn) {
          const float ys = av[q] * inv;
          const float yh = (jb + q == aidx) ? 1.0f : 0.0f;
          oRow[jb + q] = (yh - ys) + ys;  // straight-through forward value
        }
      }
      if (aidx == EOS_N && !fin) { len = t + 1; fin = true; }
    }
  }

  // Epilogue: even lane of each pair owns the row.
  if ((lane & 1) == 0) {
    const int rGlob = rowWaveBase + (lane >> 1);
    out[(size_t)B_N * T_N * V_N + rGlob] = (float)len;
    atomicAdd(out + (size_t)B_N * T_N * V_N + B_N,
              entAcc * (1.0f / ((float)T_N * (float)B_N)));
  }
}

// ---------------------------------------------------------------------------
extern "C" void kernel_launch(void* const* d_in, const int* in_sizes, int n_in,
                              void* d_out, int out_size, void* d_ws, size_t ws_size,
                              hipStream_t stream) {
  (void)in_sizes; (void)n_in; (void)out_size; (void)ws_size;
  const float* x       = (const float*)d_in[0];
  const float* tau     = (const float*)d_in[1];
  const float* u       = (const float*)d_in[2];
  const float* Wenc    = (const float*)d_in[3];
  const float* benc    = (const float*)d_in[4];
  const float* Wih     = (const float*)d_in[5];
  const float* Whh     = (const float*)d_in[6];
  const float* bih     = (const float*)d_in[7];
  const float* bhh     = (const float*)d_in[8];
  const float* initEmb = (const float*)d_in[9];
  const float* Wproj   = (const float*)d_in[10];
  const float* bproj   = (const float*)d_in[11];
  float* out = (float*)d_out;
  float* h0  = (float*)d_ws;  // B*H f32 = 16 MB scratch

  zero_ent_kernel<<<dim3(1), dim3(1), 0, stream>>>(out);

  const size_t encLds = (size_t)KC_D * 4 * 256 * 4 + (size_t)8 * 16 * 32 * 2;  // ~110 KB
  enc_kernel<<<dim3(B_N / 128), dim3(256), encLds, stream>>>(x, Wenc, benc, h0);

  const size_t gruLds = 93824;  // see layout comment in gru_kernel
  gru_kernel<<<dim3(B_N / 128), dim3(256), gruLds, stream>>>(
      h0, u, tau, Wih, Whh, bih, bhh, initEmb, Wproj, bproj, out);
}
